// LSTM_RNN_15058155339962
// MI455X (gfx1250) — compile-verified, matrix operand order checked by reading, not run on hardware
//
#include <hip/hip_runtime.h>
#include <hip/hip_bf16.h>

// ---------------------------------------------------------------------------
// Problem constants (from the reference)
// ---------------------------------------------------------------------------
#define BSZ   2048
#define TSEQ  160
#define HD    32          // LSTM hidden
#define G4    128         // 4*H
#define DFC   168
#define DFCP  176         // padded to 11 N-tiles of 16
#define NCLS  4
#define BN_N  (BSZ * DFC) // BN stat population at t = T-1

typedef __attribute__((ext_vector_type(16))) _Float16 v16h;
typedef __attribute__((ext_vector_type(8)))  float    v8f;

__device__ __forceinline__ v8f wmma_f16(v16h a, v16h b, v8f c) {
    // D = A(16x32 f16) * B(32x16 f16) + C(16x16 f32)
    return __builtin_amdgcn_wmma_f32_16x16x32_f16(
        /*neg_a=*/false, a, /*neg_b=*/false, b,
        /*c_mod=*/(short)0, c, /*reuse_a=*/false, /*reuse_b=*/false);
}

// k index for element j (0..15) of a 16-bit A/B fragment, given lane's k-half
// (CDNA5 16-bit 16x32 layout: VGPR v<4 -> K = kh*8 + 2v(+1); v>=4 -> +16)
__device__ __forceinline__ int frag_k(int j, int kh) {
    int vg = j >> 1, half = j & 1;
    return ((vg >> 2) << 4) + (kh << 3) + ((vg & 3) << 1) + half;
}

// branchless tanh: native V_TANH_F32 on gfx1250 (confirmed in round-2 asm)
__device__ __forceinline__ float fast_tanh(float x) {
#if __has_builtin(__builtin_amdgcn_tanhf)
    return __builtin_amdgcn_tanhf(x);
#elif __has_builtin(__builtin_amdgcn_exp2f) && __has_builtin(__builtin_amdgcn_rcpf)
    float e = __builtin_amdgcn_exp2f(x * 2.885390081777927f);   // exp(2x)
    return 1.0f - 2.0f * __builtin_amdgcn_rcpf(e + 1.0f);
#else
    float e = __expf(2.0f * x);
    return 1.0f - 2.0f / (e + 1.0f);
#endif
}

// quad (4-lane) broadcast, pure-VALU DPP if available
#if __has_builtin(__builtin_amdgcn_mov_dpp)
#define QUAD_BCAST(v, q)                                                      \
    __int_as_float(__builtin_amdgcn_mov_dpp(__float_as_int(v), (q)*0x55, 0xf, 0xf, true))
#else
#define QUAD_BCAST(v, q) __shfl((v), ((threadIdx.x & 31) & ~3) + (q), 32)
#endif

// ---------------------------------------------------------------------------
// Weight packing. Gate columns are PERMUTED so that packed column n holds
//   gate (n&3)  of hidden unit  hc = (n>>4)*4 + ((n>>2)&3)
// => each wave's 16-col WMMA tile = (i,f,g,o) x 4 hidden units, quads of lanes.
// ---------------------------------------------------------------------------
__global__ void pack_lstm_w_kernel(const float* __restrict__ Wih, const float* __restrict__ Whh,
                                   const float* __restrict__ bih, const float* __restrict__ bhh,
                                   int d_in,
                                   _Float16* __restrict__ wih_t, _Float16* __restrict__ whh_t,
                                   float* __restrict__ bias) {
    int idx = blockIdx.x * blockDim.x + threadIdx.x;  // 32*128
    if (idx >= 32 * G4) return;
    int k = idx / G4, n = idx % G4;
    int g  = n & 3;
    int hc = ((n >> 4) << 2) + ((n >> 2) & 3);
    int og = g * HD + hc;                 // original PyTorch gate row (i,f,g,o chunks)
    wih_t[idx] = (k < d_in) ? (_Float16)Wih[og * d_in + k] : (_Float16)0.0f;
    whh_t[idx] = (_Float16)Whh[og * HD + k];
    if (idx < G4) {
        int g2  = idx & 3;
        int hc2 = ((idx >> 4) << 2) + ((idx >> 2) & 3);
        int og2 = g2 * HD + hc2;
        bias[idx] = bih[og2] + bhh[og2];
    }
}

// W1 [168,32] -> w1t f16 [32,176] (K-major, zero pad N). No permutation.
__global__ void pack_w1_kernel(const float* __restrict__ W1, _Float16* __restrict__ w1t) {
    int idx = blockIdx.x * blockDim.x + threadIdx.x;  // 32*176
    if (idx >= 32 * DFCP) return;
    int k = idx / DFCP, n = idx % DFCP;
    w1t[idx] = (n < DFC) ? (_Float16)W1[n * HD + k] : (_Float16)0.0f;
}

// ---------------------------------------------------------------------------
// Fully fused 3-layer LSTM scan.
// Block = 256 threads = 8 waves = 16 batch rows through all layers & steps.
// Per step, per layer: 2 chained v_wmma_f32_16x16x32_f16 (B-frags & cell state
// resident in VGPRs for the whole sequence), DPP-quad gate math at full lane
// occupancy, single exec-toggle for the h stores, 1 barrier.
// h is double-buffered in LDS; nothing but x is read and only h(T-1) of layer2
// is written to global.
// ---------------------------------------------------------------------------
__global__ void __launch_bounds__(256)
lstm3_fused_kernel(const float*    __restrict__ x,     // [B, T, 6] f32
                   const _Float16* __restrict__ wih,   // [3][32][128] f16 (permuted)
                   const _Float16* __restrict__ whh,   // [3][32][128] f16 (permuted)
                   const float*    __restrict__ bias,  // [3][128] (permuted)
                   _Float16*       __restrict__ hlast) // [B, 32] f16
{
    __shared__ _Float16 h_sh[3][2][16 * HD];   // [layer][parity][row*32+col]

    const int tid  = threadIdx.x;
    const int lane = tid & 31;
    const int wv   = tid >> 5;            // 0..7 : N-tile
    const int row0 = blockIdx.x * 16;
    const int mrow = lane & 15;
    const int kh   = lane >> 4;           // K-half
    const int ncol = wv * 16 + (lane & 15);
    const int q    = (lane >> 2) & 3;     // hidden unit within wave
    const int hc   = (wv << 2) + q;       // hidden unit 0..31
    const bool owner = (lane & 3) == 0;
    const bool isG   = (lane & 3) == 2;   // this lane's column is the tanh gate

    for (int i = tid; i < 3 * 2 * 16 * HD; i += 256)
        ((_Float16*)h_sh)[i] = (_Float16)0.0f;

    // resident B fragments + bias C-init for all 3 layers
    v16h bihf[3], bhhf[3];
    v8f  cb[3];
#pragma unroll
    for (int l = 0; l < 3; ++l) {
#pragma unroll
        for (int j = 0; j < 16; ++j) {
            int k = frag_k(j, kh);
            bihf[l][j] = wih[l * 32 * G4 + k * G4 + ncol];
            bhhf[l][j] = whh[l * 32 * G4 + k * G4 + ncol];
        }
        float bv = bias[l * G4 + ncol];
#pragma unroll
        for (int r = 0; r < 8; ++r) cb[l][r] = bv;
    }

    // cell state lives in registers (replicated across each quad)
    float cst[3][8];
#pragma unroll
    for (int l = 0; l < 3; ++l)
#pragma unroll
        for (int r = 0; r < 8; ++r) cst[l][r] = 0.0f;

    // layer-0 A fragment: x padded to K=32 in-register (only kh==0, k<6 nonzero)
    const float* xrow = x + (size_t)(row0 + mrow) * TSEQ * 6;
    v16h axn;
#pragma unroll
    for (int j = 0; j < 16; ++j) axn[j] = (_Float16)0.0f;
    if (kh == 0) {
#pragma unroll
        for (int j = 0; j < 6; ++j) axn[j] = (_Float16)xrow[j];   // frag_k(j,0)==j for j<6
    }
    __syncthreads();

    for (int t = 0; t < TSEQ; ++t) {
        const int wb = t & 1, rb = (t + 1) & 1;   // write / read parity
        v16h axc = axn;
        if (t + 1 < TSEQ) {                        // prefetch next x fragment
            if (kh == 0) {
                const float* xr = xrow + (size_t)(t + 1) * 6;
#pragma unroll
                for (int j = 0; j < 6; ++j) axn[j] = (_Float16)xr[j];
            }
        }
#pragma unroll
        for (int l = 0; l < 3; ++l) {
            // A operands: layer input (x or previous layer's h at this t), own h(t-1)
            v16h a0, ah;
#pragma unroll
            for (int j = 0; j < 16; ++j) {
                int k = frag_k(j, kh);
                a0[j] = (l == 0) ? axc[j] : h_sh[l - 1][wb][mrow * HD + k];
                ah[j] = h_sh[l][rb][mrow * HD + k];
            }
            v8f d = wmma_f16(a0, bihf[l], cb[l]);   // in * Wih^T + (bih+bhh)
            d     = wmma_f16(ah, bhhf[l], d);       // + h_{t-1} * Whh^T

            // in-wave gate math (full occupancy, no divergence): quads hold
            // (i,f,g,o) of one hidden unit; results replicated across the quad
            float hh[8];
#pragma unroll
            for (int r = 0; r < 8; ++r) {
                float v  = d[r];
                float u  = isG ? v : 0.5f * v;
                float tw = fast_tanh(u);
                float act = isG ? tw : (tw * 0.5f + 0.5f);   // tanh or sigmoid
                float ia = QUAD_BCAST(act, 0);
                float fa = QUAD_BCAST(act, 1);
                float ga = QUAD_BCAST(act, 2);
                float oa = QUAD_BCAST(act, 3);
                float cc = fa * cst[l][r] + ia * ga;
                cst[l][r] = cc;
                hh[r] = oa * fast_tanh(cc);
            }
            // single exec-toggle: one lane per quad commits h
            if (owner) {
#pragma unroll
                for (int r = 0; r < 8; ++r)
                    h_sh[l][wb][(r + kh * 8) * HD + hc] = (_Float16)hh[r];
                if (l == 2 && t == TSEQ - 1) {
#pragma unroll
                    for (int r = 0; r < 8; ++r)
                        hlast[(size_t)(row0 + r + kh * 8) * HD + hc] = (_Float16)hh[r];
                }
            }
            __syncthreads();
        }
    }
}

// ---------------------------------------------------------------------------
// FC1 at t = T-1: [2048,32]f16 x [32,176]f16 + b1 -> fc [2048,168] f32
// ---------------------------------------------------------------------------
__global__ void __launch_bounds__(256)
fc1_head_kernel(const _Float16* __restrict__ hlast, // [B,32] f16
                const _Float16* __restrict__ w1t,   // [32,176] f16
                const float*    __restrict__ b1,    // [168]
                float*          __restrict__ fc)    // [2048,168]
{
    const int tid  = threadIdx.x;
    const int lane = tid & 31;
    const int wv   = tid >> 5;
    const int row0 = blockIdx.x * 16;
    const int mrow = lane & 15;
    const int kh   = lane >> 4;

    v16h a;
    const _Float16* xrow = hlast + (size_t)(row0 + mrow) * HD;
#pragma unroll
    for (int j = 0; j < 16; ++j) a[j] = xrow[frag_k(j, kh)];

    for (int nt = wv; nt < DFCP / 16; nt += 8) {      // 11 N-tiles over 8 waves
        int ncol = nt * 16 + (lane & 15);
        v16h b;
#pragma unroll
        for (int j = 0; j < 16; ++j) b[j] = w1t[frag_k(j, kh) * DFCP + ncol];
        v8f c;
        float bv = (ncol < DFC) ? b1[ncol] : 0.0f;
#pragma unroll
        for (int r = 0; r < 8; ++r) c[r] = bv;
        v8f d = wmma_f16(a, b, c);
        if (ncol < DFC) {
#pragma unroll
            for (int r = 0; r < 8; ++r)
                fc[(size_t)(row0 + r + kh * 8) * DFC + ncol] = d[r];
        }
    }
}

// ---------------------------------------------------------------------------
// BN stats over fc[2048,168]: deterministic two-level reduction
// ---------------------------------------------------------------------------
__global__ void bn_partial_kernel(const float* __restrict__ fc, float* __restrict__ part) {
    __shared__ float s1[256], s2[256];
    int tid = threadIdx.x;
    float a1 = 0.f, a2 = 0.f;
    for (size_t i = (size_t)blockIdx.x * 256 + tid; i < (size_t)BN_N; i += (size_t)gridDim.x * 256) {
        float v = fc[i];
        a1 += v; a2 += v * v;
    }
    s1[tid] = a1; s2[tid] = a2; __syncthreads();
    for (int s = 128; s > 0; s >>= 1) {
        if (tid < s) { s1[tid] += s1[tid + s]; s2[tid] += s2[tid + s]; }
        __syncthreads();
    }
    if (tid == 0) { part[blockIdx.x * 2] = s1[0]; part[blockIdx.x * 2 + 1] = s2[0]; }
}

__global__ void bn_reduce_kernel(const float* __restrict__ part, float* __restrict__ stats) {
    __shared__ float s1[256], s2[256];
    int tid = threadIdx.x;                     // 256 partials
    s1[tid] = part[tid * 2]; s2[tid] = part[tid * 2 + 1]; __syncthreads();
    for (int s = 128; s > 0; s >>= 1) {
        if (tid < s) { s1[tid] += s1[tid + s]; s2[tid] += s2[tid + s]; }
        __syncthreads();
    }
    if (tid == 0) { stats[0] = s1[0]; stats[1] = s2[0]; }
}

// ---------------------------------------------------------------------------
// BN normalize + SELU + FC2 -> logits [2048,4]
// ---------------------------------------------------------------------------
__global__ void head_final_kernel(const float* __restrict__ fc, const float* __restrict__ stats,
                                  const float* __restrict__ gamma, const float* __restrict__ beta,
                                  const float* __restrict__ W2, const float* __restrict__ b2,
                                  float* __restrict__ logits) {
    int b = blockIdx.x * blockDim.x + threadIdx.x;
    if (b >= BSZ) return;
    const float mean = stats[0] / (float)BN_N;
    const float var  = stats[1] / (float)BN_N - mean * mean;
    const float inv  = rsqrtf(var + 1e-5f) * gamma[TSEQ - 1];
    const float bet  = beta[TSEQ - 1];
    const float SC = 1.0507009873554804934f, AL = 1.6732632423543772848f;
    float acc0 = b2[0], acc1 = b2[1], acc2 = b2[2], acc3 = b2[3];
    const float* row = fc + (size_t)b * DFC;
    for (int f = 0; f < DFC; ++f) {
        float xv = (row[f] - mean) * inv + bet;
        float s = SC * (xv > 0.f ? xv : AL * (__expf(xv) - 1.f));
        acc0 += s * W2[0 * DFC + f];
        acc1 += s * W2[1 * DFC + f];
        acc2 += s * W2[2 * DFC + f];
        acc3 += s * W2[3 * DFC + f];
    }
    float* o = logits + (size_t)b * NCLS;
    o[0] = acc0; o[1] = acc1; o[2] = acc2; o[3] = acc3;
}

// ---------------------------------------------------------------------------
// softmax over the batch axis (axis=0), per class. 1 block per class.
// ---------------------------------------------------------------------------
__global__ void softmax_axis0_kernel(const float* __restrict__ logits, float* __restrict__ out) {
    __shared__ float red[256];
    const int c = blockIdx.x, tid = threadIdx.x;
    float mx = -3.402823466e38f;
    for (int i = tid; i < BSZ; i += 256) mx = fmaxf(mx, logits[i * NCLS + c]);
    red[tid] = mx; __syncthreads();
    for (int s = 128; s > 0; s >>= 1) { if (tid < s) red[tid] = fmaxf(red[tid], red[tid + s]); __syncthreads(); }
    mx = red[0]; __syncthreads();
    float sum = 0.f;
    for (int i = tid; i < BSZ; i += 256) sum += __expf(logits[i * NCLS + c] - mx);
    red[tid] = sum; __syncthreads();
    for (int s = 128; s > 0; s >>= 1) { if (tid < s) red[tid] += red[tid + s]; __syncthreads(); }
    float inv = 1.f / red[0];
    for (int i = tid; i < BSZ; i += 256) out[i * NCLS + c] = __expf(logits[i * NCLS + c] - mx) * inv;
}

// ---------------------------------------------------------------------------
// Host launcher
// ---------------------------------------------------------------------------
extern "C" void kernel_launch(void* const* d_in, const int* in_sizes, int n_in,
                              void* d_out, int out_size, void* d_ws, size_t ws_size,
                              hipStream_t stream) {
    const float* x      = (const float*)d_in[0];
    const float* Wih[3] = {(const float*)d_in[1], (const float*)d_in[5], (const float*)d_in[9]};
    const float* Whh[3] = {(const float*)d_in[2], (const float*)d_in[6], (const float*)d_in[10]};
    const float* bih[3] = {(const float*)d_in[3], (const float*)d_in[7], (const float*)d_in[11]};
    const float* bhh[3] = {(const float*)d_in[4], (const float*)d_in[8], (const float*)d_in[12]};
    const float* W1     = (const float*)d_in[13];
    const float* b1     = (const float*)d_in[14];
    const float* gamma  = (const float*)d_in[15];
    const float* beta   = (const float*)d_in[16];
    const float* W2     = (const float*)d_in[17];
    const float* b2     = (const float*)d_in[18];
    float* out = (float*)d_out;

    // workspace carve (256B aligned)
    char* w = (char*)d_ws;
    auto carve = [&](size_t bytes) { void* p = (void*)w; w += (bytes + 255) & ~(size_t)255; return p; };
    _Float16* wih_t = (_Float16*)carve(3 * 32 * G4 * sizeof(_Float16));
    _Float16* whh_t = (_Float16*)carve(3 * 32 * G4 * sizeof(_Float16));
    float*    biasp = (float*)carve(3 * G4 * sizeof(float));
    _Float16* w1t   = (_Float16*)carve(32 * DFCP * sizeof(_Float16));
    _Float16* hlast = (_Float16*)carve((size_t)BSZ * HD * sizeof(_Float16));
    float*    fc    = (float*)carve((size_t)BSZ * DFC * sizeof(float));
    float*    part  = (float*)carve(256 * 2 * sizeof(float));
    float*    stats = (float*)carve(2 * sizeof(float));
    float*    logit = (float*)carve((size_t)BSZ * NCLS * sizeof(float));

    // ---- weight packing (gate-permuted f16) ----
    for (int l = 0; l < 3; ++l) {
        pack_lstm_w_kernel<<<(32 * G4) / 256, 256, 0, stream>>>(
            Wih[l], Whh[l], bih[l], bhh[l], (l == 0 ? 6 : HD),
            wih_t + l * 32 * G4, whh_t + l * 32 * G4, biasp + l * G4);
    }
    pack_w1_kernel<<<(32 * DFCP + 255) / 256, 256, 0, stream>>>(W1, w1t);

    // ---- fused 3-layer LSTM scan (WMMA + DPP gates, h never leaves LDS) ----
    lstm3_fused_kernel<<<BSZ / 16, 256, 0, stream>>>(x, wih_t, whh_t, biasp, hlast);

    // ---- head: only t = T-1 matters (softmax is over batch; output takes last step) ----
    fc1_head_kernel<<<BSZ / 16, 256, 0, stream>>>(hlast, w1t, b1, fc);
    bn_partial_kernel<<<256, 256, 0, stream>>>(fc, part);
    bn_reduce_kernel<<<1, 256, 0, stream>>>(part, stats);
    head_final_kernel<<<BSZ / 256, 256, 0, stream>>>(fc, stats, gamma, beta, W2, b2, logit);
    softmax_axis0_kernel<<<NCLS, 256, 0, stream>>>(logit, out);
}
